// NeuralClusteringAttention_86354612453690
// MI455X (gfx1250) — matrix operand order repaired and checked by reading, not compiled, not run on hardware
//
#include <hip/hip_runtime.h>
#include <stdint.h>

// ---------------------------------------------------------------------------
// NeuralClusteringAttention for MI455X (gfx1250, wave32, WMMA bf16).
// B=4, T=1024, D=512, C=8, H=8, hd=64.
// Key identity: masked tokens have q=k=v=0 (bin_=0), so per-cluster attention
// = attention over the cluster's tokens with a (T-n)*exp(-rowmax) softmax-
// denominator correction. QKV and out-proj are done once for all tokens.
// ---------------------------------------------------------------------------

#define B_ 4
#define T_ 1024
#define D_ 512
#define C_ 8
#define H_ 8
#define HD_ 64
#define NTOK_ (B_ * T_)

typedef __bf16 bf16_t;
typedef bf16_t bf16x4  __attribute__((ext_vector_type(4)));
typedef bf16_t bf16x8  __attribute__((ext_vector_type(8)));
typedef bf16_t bf16x16 __attribute__((ext_vector_type(16)));
typedef float  f32x8   __attribute__((ext_vector_type(8)));

// A-operand (16x32 bf16, M x K): lane m=lane%16, half=lane/16.
// elems 0..7  <-> K = k0 + 16*half + i
// elems 8..15 <-> K = k0 + 16*half + 8 + i
__device__ __forceinline__ bf16x16 load_a16x32(const bf16_t* row, int half) {
  bf16x8 lo = *(const bf16x8*)(row + 16 * half);
  bf16x8 hi = *(const bf16x8*)(row + 16 * half + 8);
  return __builtin_shufflevector(lo, hi, 0, 1, 2, 3, 4, 5, 6, 7,
                                 8, 9, 10, 11, 12, 13, 14, 15);
}

// CDNA5 async copy: global -> LDS without a VGPR round-trip (ASYNCcnt).
__device__ __forceinline__ void async_g2l_b128(uint32_t lds_byte, const void* gptr) {
  asm volatile("global_load_async_to_lds_b128 %0, %1, off"
               :: "v"(lds_byte), "v"(gptr) : "memory");
}
__device__ __forceinline__ void wait_async0() {
  asm volatile("s_wait_asynccnt 0x0" ::: "memory");
}

// ---------------------------------------------------------------------------
// K0: zero the per-(batch,cluster) counters (ws is poisoned, not re-zeroed).
// ---------------------------------------------------------------------------
__global__ void k0_zero_counts(int* counts) {
  if (threadIdx.x < B_ * C_) counts[threadIdx.x] = 0;
}

// ---------------------------------------------------------------------------
// K1: cluster argmax per token + gather lists. One wave per token.
// ---------------------------------------------------------------------------
__global__ __launch_bounds__(256) void k1_assign(const float* __restrict__ X,
                                                 const float* __restrict__ Wc,
                                                 const float* __restrict__ bc,
                                                 int* __restrict__ counts,
                                                 int* __restrict__ idxl) {
  int wave = (blockIdx.x * blockDim.x + threadIdx.x) >> 5;
  int lane = threadIdx.x & 31;
  if (wave >= NTOK_) return;
  const float* x = X + (size_t)wave * D_;
  float s[C_];
#pragma unroll
  for (int c = 0; c < C_; ++c) s[c] = 0.0f;
  for (int d = lane; d < D_; d += 32) {
    float xv = x[d];
#pragma unroll
    for (int c = 0; c < C_; ++c) s[c] += xv * Wc[c * D_ + d];
  }
#pragma unroll
  for (int off = 16; off > 0; off >>= 1) {
#pragma unroll
    for (int c = 0; c < C_; ++c) s[c] += __shfl_xor(s[c], off, 32);
  }
  if (lane == 0) {
    int best = 0;
    float bv = s[0] + bc[0];
#pragma unroll
    for (int c = 1; c < C_; ++c) {
      float v = s[c] + bc[c];
      if (v > bv) { bv = v; best = c; }  // strict >: first max wins (argmax)
    }
    int b = wave >> 10;            // token/T
    int tl = wave & (T_ - 1);      // token within batch
    int slot = b * C_ + best;
    int pos = atomicAdd(&counts[slot], 1);
    idxl[slot * T_ + pos] = tl;
  }
}

// ---------------------------------------------------------------------------
// K2: fp32 -> bf16 conversion of X, Win, Wout (vectorized, 4 elems/thread).
// ---------------------------------------------------------------------------
__global__ void k2_convert(const float* __restrict__ X,
                           const float* __restrict__ Win,
                           const float* __restrict__ Wout,
                           bf16_t* __restrict__ Xbf,
                           bf16_t* __restrict__ Winbf,
                           bf16_t* __restrict__ Woutbf) {
  size_t i = ((size_t)blockIdx.x * blockDim.x + threadIdx.x) * 4;
  const size_t nx = (size_t)NTOK_ * D_;       // 2,097,152
  const size_t nw = (size_t)3 * D_ * D_;      //   786,432
  const size_t no = (size_t)D_ * D_;          //   262,144
  if (i < nx) {
    float4 v = *(const float4*)(X + i);
    bf16x4 o = {(bf16_t)v.x, (bf16_t)v.y, (bf16_t)v.z, (bf16_t)v.w};
    *(bf16x4*)(Xbf + i) = o;
  }
  if (i < nw) {
    float4 v = *(const float4*)(Win + i);
    bf16x4 o = {(bf16_t)v.x, (bf16_t)v.y, (bf16_t)v.z, (bf16_t)v.w};
    *(bf16x4*)(Winbf + i) = o;
  }
  if (i < no) {
    float4 v = *(const float4*)(Wout + i);
    bf16x4 o = {(bf16_t)v.x, (bf16_t)v.y, (bf16_t)v.z, (bf16_t)v.w};
    *(bf16x4*)(Woutbf + i) = o;
  }
}

// ---------------------------------------------------------------------------
// K3: QKV GEMM. qkv[t,j] = sum_d Xbf[t,d]*Winbf[j,d] + bin[j].
// One wave per 16x64 output block: A loaded once per k-step; all four B
// operands are preloaded before the WMMA burst so waits can stagger.
// ---------------------------------------------------------------------------
__global__ __launch_bounds__(256) void k3_qkv(const bf16_t* __restrict__ Xbf,
                                              const bf16_t* __restrict__ Winbf,
                                              const float* __restrict__ bin,
                                              bf16_t* __restrict__ Q,
                                              bf16_t* __restrict__ K,
                                              bf16_t* __restrict__ V) {
  const int NG = (3 * D_) / 64;  // 24 column groups of 64
  int wave = (blockIdx.x * blockDim.x + threadIdx.x) >> 5;
  int lane = threadIdx.x & 31;
  int mt = wave / NG, ng = wave % NG;
  int m = lane & 15, half = lane >> 4;

  const bf16_t* arow = Xbf + (size_t)(mt * 16 + m) * D_;
  const bf16_t* brow[4];
#pragma unroll
  for (int sub = 0; sub < 4; ++sub)
    brow[sub] = Winbf + (size_t)(ng * 64 + sub * 16 + m) * D_;

  f32x8 acc[4];
#pragma unroll
  for (int sub = 0; sub < 4; ++sub) acc[sub] = (f32x8){};

#pragma unroll 2
  for (int k0 = 0; k0 < D_; k0 += 32) {
    __builtin_prefetch(arow + k0 + 64, 0, 3);  // WGP-scope prefetch, next chunk
    bf16x16 a = load_a16x32(arow + k0, half);
    bf16x16 bb[4];
#pragma unroll
    for (int sub = 0; sub < 4; ++sub)
      bb[sub] = *(const bf16x16*)(brow[sub] + k0 + 16 * half);
#pragma unroll
    for (int sub = 0; sub < 4; ++sub)
      acc[sub] = __builtin_amdgcn_wmma_f32_16x16x32_bf16(false, a, false, bb[sub],
                                                         (short)0, acc[sub], false, false);
  }

  int jbase = ng * 64;
  int which = jbase >> 9;  // 0=Q 1=K 2=V (64-col groups never straddle)
  bf16_t* dst = (which == 0) ? Q : ((which == 1) ? K : V);
  int jmb = jbase & (D_ - 1);
#pragma unroll
  for (int sub = 0; sub < 4; ++sub) {
    float bias = bin[jbase + sub * 16 + m];
    int jm = jmb + sub * 16 + m;
#pragma unroll
    for (int r = 0; r < 8; ++r) {
      int tok = mt * 16 + r + 8 * half;
      dst[(size_t)tok * D_ + jm] = (bf16_t)(acc[sub][r] + bias);
    }
  }
}

// ---------------------------------------------------------------------------
// K4: gathered per-(b,c) flash attention, one wave per (qtile, head, b*c).
// Softmax over all T keys = in-cluster keys + (T-n) implicit zero-score keys.
// V tile staged via GLOBAL_LOAD_ASYNC_TO_LDS (overlaps with score WMMAs).
// ---------------------------------------------------------------------------
__global__ __launch_bounds__(32) void k4_attn(const bf16_t* __restrict__ Qbf,
                                              const bf16_t* __restrict__ Kbf,
                                              const bf16_t* __restrict__ Vbf,
                                              const int* __restrict__ counts,
                                              const int* __restrict__ idxl,
                                              bf16_t* __restrict__ Obf) {
  int qt = blockIdx.x;   // query tile (0..63)
  int h = blockIdx.y;    // head
  int bc = blockIdx.z;   // b*C + c
  int n = counts[bc];
  if (qt * 16 >= n) return;  // uniform: whole wave exits, EXEC stays full
  int b = bc >> 3;
  const int* idx = idxl + bc * T_;

  int lane = threadIdx.x;
  int m = lane & 15, half = lane >> 4;

  __shared__ __align__(16) bf16_t pbuf[16 * 32];   // P tile: 16 q rows x 32 keys
  __shared__ __align__(16) bf16_t vtile[32 * 64];  // V tile: 32 keys x hd

  // Query A-operands (reused over the whole key loop)
  int qrow = qt * 16 + m;
  int qcl = (qrow < n) ? qrow : (n - 1);
  const bf16_t* qptr = Qbf + ((size_t)(b * T_ + idx[qcl]) * D_) + h * HD_;
  bf16x16 a0 = load_a16x32(qptr, half);       // hd K 0..31
  bf16x16 a1 = load_a16x32(qptr + 32, half);  // hd K 32..63

  bool hasOut = (n < T_);
  float rowmax[8], rowsum[8];
#pragma unroll
  for (int r = 0; r < 8; ++r) {
    rowmax[r] = hasOut ? 0.0f : -1e30f;  // implicit zero-score keys seed the max
    rowsum[r] = 0.0f;
  }
  f32x8 oc[4];
#pragma unroll
  for (int nc = 0; nc < 4; ++nc) oc[nc] = (f32x8){};

  for (int kt = 0; kt < n; kt += 32) {
    // Async-stage 32 V rows into LDS (one row per lane, 8 x b128 per lane).
    {
      int kr = kt + lane;
      int kcl = (kr < n) ? kr : (n - 1);
      const bf16_t* vptr = Vbf + ((size_t)(b * T_ + idx[kcl]) * D_) + h * HD_;
      uint32_t vdst = (uint32_t)(uintptr_t)(vtile + (size_t)lane * HD_);
#pragma unroll
      for (int j = 0; j < 8; ++j)
        async_g2l_b128(vdst + j * 16, vptr + j * 8);
    }

    // Scores: two 16x16 tiles (keys kt..kt+15, kt+16..kt+31), K-dim = hd = 64.
    // These global loads + 4 WMMAs overlap the async V copy above.
    f32x8 s0 = {}, s1 = {};
    int kcol = kt + m;
    {
      int c0 = (kcol < n) ? kcol : (n - 1);
      int c1 = (kcol + 16 < n) ? (kcol + 16) : (n - 1);
      const bf16_t* kp0 = Kbf + ((size_t)(b * T_ + idx[c0]) * D_) + h * HD_;
      const bf16_t* kp1 = Kbf + ((size_t)(b * T_ + idx[c1]) * D_) + h * HD_;
      bf16x16 b00 = *(const bf16x16*)(kp0 + 16 * half);
      bf16x16 b01 = *(const bf16x16*)(kp0 + 32 + 16 * half);
      bf16x16 b10 = *(const bf16x16*)(kp1 + 16 * half);
      bf16x16 b11 = *(const bf16x16*)(kp1 + 32 + 16 * half);
      s0 = __builtin_amdgcn_wmma_f32_16x16x32_bf16(false, a0, false, b00, (short)0, s0, false, false);
      s0 = __builtin_amdgcn_wmma_f32_16x16x32_bf16(false, a1, false, b01, (short)0, s0, false, false);
      s1 = __builtin_amdgcn_wmma_f32_16x16x32_bf16(false, a0, false, b10, (short)0, s1, false, false);
      s1 = __builtin_amdgcn_wmma_f32_16x16x32_bf16(false, a1, false, b11, (short)0, s1, false, false);
    }
    bool ok0 = (kcol < n);
    bool ok1 = (kcol + 16 < n);
#pragma unroll
    for (int r = 0; r < 8; ++r) {
      s0[r] = ok0 ? s0[r] * 0.125f : -1e30f;  // 1/sqrt(64)
      s1[r] = ok1 ? s1[r] * 0.125f : -1e30f;
    }

    // Row max across the 16-lane half (rows live per-VGPR within a half).
    float nm[8];
#pragma unroll
    for (int r = 0; r < 8; ++r) nm[r] = fmaxf(s0[r], s1[r]);
#pragma unroll
    for (int off = 1; off < 16; off <<= 1) {
#pragma unroll
      for (int r = 0; r < 8; ++r) nm[r] = fmaxf(nm[r], __shfl_xor(nm[r], off, 32));
    }
    float alpha[8];
#pragma unroll
    for (int r = 0; r < 8; ++r) {
      float newm = fmaxf(rowmax[r], nm[r]);
      alpha[r] = __expf(rowmax[r] - newm);
      rowmax[r] = newm;
    }
    float p0[8], p1[8], psum[8];
#pragma unroll
    for (int r = 0; r < 8; ++r) {
      p0[r] = __expf(s0[r] - rowmax[r]);
      p1[r] = __expf(s1[r] - rowmax[r]);
      psum[r] = p0[r] + p1[r];
    }
#pragma unroll
    for (int off = 1; off < 16; off <<= 1) {
#pragma unroll
      for (int r = 0; r < 8; ++r) psum[r] += __shfl_xor(psum[r], off, 32);
    }
#pragma unroll
    for (int r = 0; r < 8; ++r) rowsum[r] = rowsum[r] * alpha[r] + psum[r];
#pragma unroll
    for (int nc = 0; nc < 4; ++nc) {
#pragma unroll
      for (int r = 0; r < 8; ++r) oc[nc][r] = oc[nc][r] * alpha[r];
    }

    // P (C-layout f32) -> LDS -> A-layout bf16 (16 rows x K=32 keys).
#pragma unroll
    for (int r = 0; r < 8; ++r) {
      int mm = r + 8 * half;
      pbuf[mm * 32 + m] = (bf16_t)p0[r];
      pbuf[mm * 32 + 16 + m] = (bf16_t)p1[r];
    }
    __builtin_amdgcn_s_barrier();  // single-wave block: S_NOP, ordering aid only
    bf16x16 pa = load_a16x32(pbuf + m * 32, half);

    wait_async0();  // V tile resident in LDS before B-operand gather

    // O += P(16x32) @ V(32x64): 4 WMMAs over hd column chunks.
#pragma unroll
    for (int nc = 0; nc < 4; ++nc) {
      bf16x16 vb;
      int colhd = nc * 16 + m;
#pragma unroll
      for (int i = 0; i < 16; ++i) vb[i] = vtile[(16 * half + i) * HD_ + colhd];
      oc[nc] = __builtin_amdgcn_wmma_f32_16x16x32_bf16(false, pa, false, vb,
                                                       (short)0, oc[nc], false, false);
    }
    __builtin_amdgcn_s_barrier();
  }

  // Normalize: Z = rowsum + (T-n)*exp(0 - rowmax); scatter to Obf.
  float extra = hasOut ? (float)(T_ - n) : 0.0f;
#pragma unroll
  for (int r = 0; r < 8; ++r) {
    float z = rowsum[r] + extra * __expf(-rowmax[r]);
    float inv = 1.0f / z;
    int qr = qt * 16 + r + 8 * half;
    if (qr < n) {
      bf16_t* optr = Obf + ((size_t)(b * T_ + idx[qr]) * D_) + h * HD_;
#pragma unroll
      for (int nc = 0; nc < 4; ++nc) optr[nc * 16 + m] = (bf16_t)(oc[nc][r] * inv);
    }
  }
}

// ---------------------------------------------------------------------------
// K5: out-projection GEMM. out[t,j] = sum_d Obf[t,d]*Woutbf[j,d] + bout[j].
// One wave per 16x64 output block, B operands preloaded per k-step.
// ---------------------------------------------------------------------------
__global__ __launch_bounds__(256) void k5_outproj(const bf16_t* __restrict__ Obf,
                                                  const bf16_t* __restrict__ Woutbf,
                                                  const float* __restrict__ bout,
                                                  float* __restrict__ out) {
  const int NG = D_ / 64;  // 8 column groups
  int wave = (blockIdx.x * blockDim.x + threadIdx.x) >> 5;
  int lane = threadIdx.x & 31;
  int mt = wave / NG, ng = wave % NG;
  int m = lane & 15, half = lane >> 4;

  const bf16_t* arow = Obf + (size_t)(mt * 16 + m) * D_;
  const bf16_t* brow[4];
#pragma unroll
  for (int sub = 0; sub < 4; ++sub)
    brow[sub] = Woutbf + (size_t)(ng * 64 + sub * 16 + m) * D_;

  f32x8 acc[4];
#pragma unroll
  for (int sub = 0; sub < 4; ++sub) acc[sub] = (f32x8){};

#pragma unroll 2
  for (int k0 = 0; k0 < D_; k0 += 32) {
    __builtin_prefetch(arow + k0 + 64, 0, 3);
    bf16x16 a = load_a16x32(arow + k0, half);
    bf16x16 bb[4];
#pragma unroll
    for (int sub = 0; sub < 4; ++sub)
      bb[sub] = *(const bf16x16*)(brow[sub] + k0 + 16 * half);
#pragma unroll
    for (int sub = 0; sub < 4; ++sub)
      acc[sub] = __builtin_amdgcn_wmma_f32_16x16x32_bf16(false, a, false, bb[sub],
                                                         (short)0, acc[sub], false, false);
  }
#pragma unroll
  for (int sub = 0; sub < 4; ++sub) {
    int j = ng * 64 + sub * 16 + m;
    float bias = bout[j];
#pragma unroll
    for (int r = 0; r < 8; ++r) {
      int tok = mt * 16 + r + 8 * half;
      out[(size_t)tok * D_ + j] = acc[sub][r] + bias;
    }
  }
}

// ---------------------------------------------------------------------------
// Host launcher.
// ---------------------------------------------------------------------------
extern "C" void kernel_launch(void* const* d_in, const int* in_sizes, int n_in,
                              void* d_out, int out_size, void* d_ws, size_t ws_size,
                              hipStream_t stream) {
  (void)in_sizes; (void)n_in; (void)out_size; (void)ws_size;
  const float* X = (const float*)d_in[0];
  const float* Wc = (const float*)d_in[1];
  const float* bc = (const float*)d_in[2];
  const float* Win = (const float*)d_in[3];
  const float* bin = (const float*)d_in[4];
  const float* Wout = (const float*)d_in[5];
  const float* bout = (const float*)d_in[6];
  float* out = (float*)d_out;

  size_t off = 0;
  auto take = [&](size_t bytes) -> void* {
    void* p = (char*)d_ws + off;
    off += (bytes + 255) & ~(size_t)255;
    return p;
  };
  int* counts = (int*)take((size_t)B_ * C_ * sizeof(int));
  int* idxl = (int*)take((size_t)B_ * C_ * T_ * sizeof(int));
  bf16_t* Xbf = (bf16_t*)take((size_t)NTOK_ * D_ * 2);
  bf16_t* Winbf = (bf16_t*)take((size_t)3 * D_ * D_ * 2);
  bf16_t* Woutbf = (bf16_t*)take((size_t)D_ * D_ * 2);
  bf16_t* Qbf = (bf16_t*)take((size_t)NTOK_ * D_ * 2);
  bf16_t* Kbf = (bf16_t*)take((size_t)NTOK_ * D_ * 2);
  bf16_t* Vbf = (bf16_t*)take((size_t)NTOK_ * D_ * 2);
  bf16_t* Obf = (bf16_t*)take((size_t)NTOK_ * D_ * 2);

  k0_zero_counts<<<1, 64, 0, stream>>>(counts);
  k1_assign<<<(NTOK_ * 32) / 256, 256, 0, stream>>>(X, Wc, bc, counts, idxl);
  k2_convert<<<2048, 256, 0, stream>>>(X, Win, Wout, Xbf, Winbf, Woutbf);
  // 256 m-tiles * 24 col-groups = 6144 waves, 8 waves/block
  k3_qkv<<<768, 256, 0, stream>>>(Xbf, Winbf, bin, Qbf, Kbf, Vbf);
  k4_attn<<<dim3(T_ / 16, H_, B_ * C_), 32, 0, stream>>>(Qbf, Kbf, Vbf, counts, idxl, Obf);
  // 256 m-tiles * 8 col-groups = 2048 waves, 8 waves/block
  k5_outproj<<<256, 256, 0, stream>>>(Obf, Woutbf, bout, out);
}
